// EIGTower_18098992185793
// MI455X (gfx1250) — matrix operand order repaired and checked by reading, not compiled, not run on hardware
//
#include <hip/hip_runtime.h>
#include <cstdint>
#include <cstddef>

#define N_NODES 30000
#define N_EDGES 480000
#define IN_F    128
#define OUT_F   128
#define ED_F    16
#define AVG_DL  2.833f
#define EPSV    1e-5f

typedef __bf16 bf16_t;
typedef __attribute__((ext_vector_type(16))) bf16_t v16bf;
typedef __attribute__((ext_vector_type(8)))  bf16_t bf16x8;
typedef __attribute__((ext_vector_type(8)))  float  v8f;
typedef __attribute__((ext_vector_type(4)))  unsigned int u32x4;
typedef __attribute__((ext_vector_type(8)))  int i32x8;
typedef __attribute__((ext_vector_type(4)))  int i32x4;

// ---------------- float atomic max/min via sign-aware integer atomics ----------------
__device__ __forceinline__ void atomicMaxFloat(float* addr, float v) {
  if (v >= 0.0f) atomicMax((int*)addr, __float_as_int(v));
  else           atomicMin((unsigned int*)addr, __float_as_uint(v));
}
__device__ __forceinline__ void atomicMinFloat(float* addr, float v) {
  if (v >= 0.0f) atomicMin((int*)addr, __float_as_int(v));
  else           atomicMax((unsigned int*)addr, __float_as_uint(v));
}

// ---------------- Tensor Data Mover: 1-D tile copy global -> LDS ----------------
// Builds a D# per CDNA5 ISA ch.8: group0 = {count,lds_addr,global_addr,type=2},
// group1 = {data_size=8B, tensor_dim0/tile_dim0 = nbytes/8, tensor_dim1=tile_dim1=1}.
// This toolchain exposes the 6-arg builtin: (g0, g1, g2, g3, g4, cpol).
__device__ __forceinline__ unsigned lds_offset_of(const void* p) {
  // flat shared-aperture address: low 32 bits are the LDS byte offset
  return (unsigned)(unsigned long long)(uintptr_t)p;
}

__device__ __forceinline__ void tdm_load_1d(unsigned lds_off, const void* gsrc,
                                            unsigned nbytes) {
  unsigned long long ga = (unsigned long long)(uintptr_t)gsrc;
  unsigned n8 = nbytes >> 3;  // 8-byte elements
  u32x4 g0;
  g0[0] = 1u;                                                // count=1 (valid user D#)
  g0[1] = lds_off;                                           // lds_addr [63:32]
  g0[2] = (unsigned)(ga & 0xFFFFFFFFu);                      // global_addr lo
  g0[3] = (unsigned)((ga >> 32) & 0x01FFFFFFu) | (2u << 30); // addr[56:32], type=2

  // group1 assembled as 4 x 64-bit words per the bitfield table
  unsigned long long q0 = ((unsigned long long)3u << 16)                 // data_size = 8B
                        | ((unsigned long long)(n8 & 0xFFFFu) << 48);    // tensor_dim0[15:0]
  unsigned long long q1 = ((unsigned long long)((n8 >> 16) & 0xFFFFu))   // tensor_dim0[31:16]
                        | ((unsigned long long)1u << 16)                 // tensor_dim1 = 1
                        | ((unsigned long long)(n8 & 0xFFFFu) << 48);    // tile_dim0
  unsigned long long q2 = 1ull                                           // tile_dim1 = 1
                        | ((unsigned long long)n8 << 32);                // tensor_dim0_stride lo
  unsigned long long q3 = 0ull;                                          // stride hi / dim1_stride

  i32x8 g1;
  g1[0] = (int)(unsigned)(q0 & 0xFFFFFFFFu); g1[1] = (int)(unsigned)(q0 >> 32);
  g1[2] = (int)(unsigned)(q1 & 0xFFFFFFFFu); g1[3] = (int)(unsigned)(q1 >> 32);
  g1[4] = (int)(unsigned)(q2 & 0xFFFFFFFFu); g1[5] = (int)(unsigned)(q2 >> 32);
  g1[6] = (int)(unsigned)(q3 & 0xFFFFFFFFu); g1[7] = (int)(unsigned)(q3 >> 32);
  i32x4 g2 = {0, 0, 0, 0};
  i32x4 g3 = {0, 0, 0, 0};
  i32x8 g4 = {0, 0, 0, 0, 0, 0, 0, 0};
  __builtin_amdgcn_tensor_load_to_lds(g0, g1, g2, g3, g4, 0);
}

// ---------------- init accumulators (re-run every launch: deterministic) ----------------
__global__ void k_init(float* sum, float* sumsq, float* el, float* mx, float* mn,
                       float* deg, float* colsum, float* colsq) {
  int i = blockIdx.x * blockDim.x + threadIdx.x;
  if (i >= N_NODES * IN_F) return;
  sum[i] = 0.0f; sumsq[i] = 0.0f; el[i] = 0.0f;
  mx[i] = -__builtin_inff();
  mn[i] =  __builtin_inff();
  if (i < N_NODES) deg[i] = 0.0f;
  if (i < OUT_F) { colsum[i] = 0.0f; colsq[i] = 0.0f; }
}

// ---------------- f32 -> bf16 conversion ----------------
__global__ void k_cvt(bf16_t* dst, const float* src, int n) {
  int i = blockIdx.x * blockDim.x + threadIdx.x;
  if (i < n) dst[i] = (bf16_t)src[i];
}

// ---------------- pre-swizzle weights into WMMA B-fragment order ----------------
__global__ void k_wfrag(bf16_t* dstF, const float* W, int kmax, int nchunks) {
  int t = blockIdx.x * blockDim.x + threadIdx.x;
  int total = nchunks * 8 * 32 * 16;
  if (t >= total) return;
  int i    = t & 15;
  int lane = (t >> 4) & 31;
  int frag = t >> 9;
  int kc = frag >> 3;
  int n  = frag & 7;
  int Klocal = ((i < 8) ? i : (i + 8)) + ((lane < 16) ? 0 : 8);
  int Kg = kc * 32 + Klocal;
  int Ncol = n * 16 + (lane & 15);
  float v = (Kg < kmax) ? W[(size_t)Kg * 128 + Ncol] : 0.0f;
  dstF[t] = (bf16_t)v;
}

// ---------------- edge filter MLP (6->3->3->1, sigmoid) + degree count ----------------
__global__ void k_edgefilt(const float* __restrict__ eig, const int* __restrict__ src,
                           const int* __restrict__ dst,
                           const float* W1, const float* b1,
                           const float* W2, const float* b2,
                           const float* W3, const float* b3,
                           float* wl, float* deg) {
  int e = blockIdx.x * blockDim.x + threadIdx.x;
  if (e >= N_EDGES) return;
  int s = src[e], d = dst[e];
  float x[6];
  x[0] = eig[s * 4 + 1]; x[1] = eig[s * 4 + 2]; x[2] = eig[s * 4 + 3];
  x[3] = eig[d * 4 + 1]; x[4] = eig[d * 4 + 2]; x[5] = eig[d * 4 + 3];
  float h1[3];
  #pragma unroll
  for (int j = 0; j < 3; j++) {
    float a = b1[j];
    #pragma unroll
    for (int k = 0; k < 6; k++) a += x[k] * W1[k * 3 + j];
    h1[j] = fmaxf(a, 0.0f);
  }
  float h2[3];
  #pragma unroll
  for (int j = 0; j < 3; j++) {
    float a = b2[j];
    #pragma unroll
    for (int k = 0; k < 3; k++) a += h1[k] * W2[k * 3 + j];
    h2[j] = fmaxf(a, 0.0f);
  }
  float o = b3[0];
  #pragma unroll
  for (int k = 0; k < 3; k++) o += h2[k] * W3[k];
  wl[e] = 1.0f / (1.0f + __expf(-o));
  unsafeAtomicAdd(&deg[d], 1.0f);
}

// ---------------- A-fragment span gather for the edge GEMM ----------------
__device__ __forceinline__ bf16x8 load_span_pre(const bf16_t* __restrict__ hb,
                                                const bf16_t* __restrict__ efb,
                                                int s, int d, int e, int K) {
  if (K < 128) return *(const bf16x8*)(hb + (size_t)s * 128 + K);
  if (K < 256) return *(const bf16x8*)(hb + (size_t)d * 128 + (K - 128));
  if (K < 272) return *(const bf16x8*)(efb + (size_t)e * 16 + (K - 256));
  bf16x8 z;
  #pragma unroll
  for (int i = 0; i < 8; i++) z[i] = (bf16_t)0.0f;
  return z;
}

// ---------------- edge GEMM (E x 272 @ 272 x 128) + scatter aggregation ----------------
// B operand (full pre-swizzled W_pre, 72KB) staged to LDS once per block via TDM.
__global__ __launch_bounds__(256) void k_edge_gemm(
    const bf16_t* __restrict__ hb, const bf16_t* __restrict__ efb,
    const bf16_t* __restrict__ WpreF, const float* __restrict__ b_pre,
    const int* __restrict__ src, const int* __restrict__ dst,
    const float* __restrict__ wl,
    float* sum, float* sumsq, float* el, float* mx, float* mn) {
  __shared__ __align__(32) bf16_t Bs[9 * 8 * 32 * 16];   // 73728 B

  int tid  = threadIdx.x;
  int lane = tid & 31;

  if ((tid >> 5) == 0) {                    // wave 0 issues the TDM DMA
    tdm_load_1d(lds_offset_of(Bs), WpreF, 9 * 8 * 32 * 16 * 2);
    __builtin_amdgcn_s_wait_tensorcnt(0);
  }
  __syncthreads();

  int wid  = (blockIdx.x * blockDim.x + tid) >> 5;   // one wave = 16 edges
  int m0   = wid * 16;                                // E multiple of 16
  int erow = m0 + (lane & 15);
  int s = src[erow], d = dst[erow];

  v8f acc[8];
  #pragma unroll
  for (int i = 0; i < 8; i++)
    #pragma unroll
    for (int j = 0; j < 8; j++) acc[i][j] = 0.0f;

  int koff = (lane >= 16) ? 8 : 0;
  bf16x8 s0 = load_span_pre(hb, efb, s, d, erow, koff);
  bf16x8 s1 = load_span_pre(hb, efb, s, d, erow, koff + 16);

  #pragma unroll
  for (int kc = 0; kc < 9; kc++) {
    v16bf a;
    #pragma unroll
    for (int i = 0; i < 8; i++) { a[i] = s0[i]; a[8 + i] = s1[i]; }
    if (kc < 8) {                         // prefetch next chunk's A spans
      int K0 = (kc + 1) * 32 + koff;
      s0 = load_span_pre(hb, efb, s, d, erow, K0);
      s1 = load_span_pre(hb, efb, s, d, erow, K0 + 16);
    }
    #pragma unroll
    for (int n = 0; n < 8; n++) {
      v16bf b = *(const v16bf*)(Bs + (size_t)((kc * 8 + n) * 32 + lane) * 16);
      acc[n] = __builtin_amdgcn_wmma_f32_16x16x32_bf16(
          false, a, false, b, (short)0, acc[n], false, false);
    }
  }

  int fl = lane & 15;
  float bp[8];
  #pragma unroll
  for (int n = 0; n < 8; n++) bp[n] = b_pre[n * 16 + fl];

  int moff = (lane >= 16) ? 8 : 0;
  #pragma unroll
  for (int r = 0; r < 8; r++) {
    int eidx = m0 + moff + r;
    int dd = dst[eidx];
    float w = wl[eidx];
    size_t base = (size_t)dd * 128;
    #pragma unroll
    for (int n = 0; n < 8; n++) {
      float v = acc[n][r] + bp[n];
      size_t off = base + n * 16 + fl;
      unsafeAtomicAdd(&sum[off], v);
      unsafeAtomicAdd(&sumsq[off], v * v);
      unsafeAtomicAdd(&el[off], v * w);
      atomicMaxFloat(&mx[off], v);
      atomicMinFloat(&mn[off], v);
    }
  }
}

// ---------------- build bf16 node-GEMM input row: [h | agg | agg*l1 | agg*l2] ----------------
__global__ void k_build_xb(const float* __restrict__ h, const float* __restrict__ deg,
                           const float* __restrict__ sum, const float* __restrict__ sumsq,
                           const float* __restrict__ mx, const float* __restrict__ mn,
                           const float* __restrict__ el, bf16_t* __restrict__ xb) {
  int idx = blockIdx.x * blockDim.x + threadIdx.x;
  if (idx >= N_NODES * IN_F) return;
  int node = idx >> 7, f = idx & 127;
  float dg = deg[node];
  float mask = (dg > 0.0f) ? 1.0f : 0.0f;
  float sd = fmaxf(dg, 1.0f);
  float mean = sum[idx] / sd;
  float msq  = sumsq[idx] / sd;
  float stdv = sqrtf(fmaxf(msq - mean * mean, 0.0f) + EPSV);
  float mxv = (dg > 0.0f) ? mx[idx] : 0.0f;
  float mnv = (dg > 0.0f) ? mn[idx] : 0.0f;
  float elv = el[idx];
  float logd = logf(dg + 1.0f);
  float l1 = logd * (1.0f / AVG_DL);
  float l2 = AVG_DL / ((logd > 0.0f) ? logd : 1.0f);
  bf16_t* row = xb + (size_t)node * 2048;
  row[f] = (bf16_t)h[idx];
  float ag[5] = { mean * mask, mxv, mnv, stdv * mask, elv * mask };
  #pragma unroll
  for (int g = 0; g < 5; g++) {
    row[128  + g * 128 + f] = (bf16_t)(ag[g]);
    row[768  + g * 128 + f] = (bf16_t)(ag[g] * l1);
    row[1408 + g * 128 + f] = (bf16_t)(ag[g] * l2);
  }
}

// ---------------- node GEMM (N x 2048 @ 2048 x 128) + snorm + column stats ----------------
// B chunks (8KB each) double-buffered in LDS via TDM; A spans software-pipelined.
__global__ __launch_bounds__(256) void k_node_gemm(
    const bf16_t* __restrict__ xb, const bf16_t* __restrict__ WpostF,
    const float* __restrict__ b_post, const float* __restrict__ snorm,
    float* __restrict__ outpre, float* colsum, float* colsq) {
  __shared__ float ls[OUT_F];
  __shared__ float lq[OUT_F];
  __shared__ __align__(32) bf16_t Bs[2][8 * 32 * 16];   // 2 x 8KB

  int tid = threadIdx.x;
  if (tid < OUT_F) { ls[tid] = 0.0f; lq[tid] = 0.0f; }

  int lane = tid & 31;
  int wib  = tid >> 5;
  int wid  = blockIdx.x * 8 + wib;
  int m0   = wid * 16;
  bool active = (m0 < N_NODES);

  if (wib == 0) {                       // prime chunk 0
    tdm_load_1d(lds_offset_of(&Bs[0][0]), WpostF, 8 * 32 * 16 * 2);
    __builtin_amdgcn_s_wait_tensorcnt(0);
  }
  __syncthreads();

  v8f acc[8];
  #pragma unroll
  for (int i = 0; i < 8; i++)
    #pragma unroll
    for (int j = 0; j < 8; j++) acc[i][j] = 0.0f;

  const bf16_t* rowA = xb + (size_t)(active ? (m0 + (lane & 15)) : 0) * 2048;
  int koff = (lane >= 16) ? 8 : 0;
  bf16x8 s0 = *(const bf16x8*)(rowA + koff);
  bf16x8 s1 = *(const bf16x8*)(rowA + koff + 16);

  for (int kc = 0; kc < 64; kc++) {
    int cur = kc & 1;
    if (wib == 0 && kc + 1 < 64)        // DMA next B chunk while computing this one
      tdm_load_1d(lds_offset_of(&Bs[cur ^ 1][0]),
                  WpostF + (size_t)(kc + 1) * (8 * 32 * 16), 8 * 32 * 16 * 2);
    v16bf a;
    #pragma unroll
    for (int i = 0; i < 8; i++) { a[i] = s0[i]; a[8 + i] = s1[i]; }
    if (kc + 1 < 64) {                  // prefetch next A spans
      int K0 = (kc + 1) * 32 + koff;
      s0 = *(const bf16x8*)(rowA + K0);
      s1 = *(const bf16x8*)(rowA + K0 + 16);
    }
    if (active) {
      #pragma unroll
      for (int n = 0; n < 8; n++) {
        v16bf b = *(const v16bf*)(&Bs[cur][(size_t)(n * 32 + lane) * 16]);
        acc[n] = __builtin_amdgcn_wmma_f32_16x16x32_bf16(
            false, a, false, b, (short)0, acc[n], false, false);
      }
    }
    if (kc + 1 < 64) {
      if (wib == 0) __builtin_amdgcn_s_wait_tensorcnt(0);
      __syncthreads();
    }
  }

  if (active) {
    int fl = lane & 15;
    float bp[8];
    #pragma unroll
    for (int n = 0; n < 8; n++) bp[n] = b_post[n * 16 + fl];
    int moff = (lane >= 16) ? 8 : 0;
    #pragma unroll
    for (int r = 0; r < 8; r++) {
      int node = m0 + moff + r;
      float sn = snorm[node];
      #pragma unroll
      for (int n = 0; n < 8; n++) {
        float v = (acc[n][r] + bp[n]) * sn;
        outpre[(size_t)node * 128 + n * 16 + fl] = v;
        atomicAdd(&ls[n * 16 + fl], v);
        atomicAdd(&lq[n * 16 + fl], v * v);
      }
    }
  }
  __syncthreads();
  if (tid < OUT_F) {
    unsafeAtomicAdd(&colsum[tid], ls[tid]);
    unsafeAtomicAdd(&colsq[tid], lq[tid]);
  }
}

// ---------------- batch-norm finalize ----------------
__global__ void k_bn(const float* __restrict__ outpre, const float* __restrict__ colsum,
                     const float* __restrict__ colsq, const float* __restrict__ gamma,
                     const float* __restrict__ beta, float* __restrict__ out) {
  int idx = blockIdx.x * blockDim.x + threadIdx.x;
  if (idx >= N_NODES * OUT_F) return;
  int f = idx & 127;
  const float invN = 1.0f / (float)N_NODES;
  float mu = colsum[f] * invN;
  float var = colsq[f] * invN - mu * mu;
  out[idx] = (outpre[idx] - mu) * rsqrtf(var + 1e-5f) * gamma[f] + beta[f];
}

// ---------------- host launch ----------------
static inline size_t align_up(size_t x) { return (x + 255) & ~(size_t)255; }

extern "C" void kernel_launch(void* const* d_in, const int* in_sizes, int n_in,
                              void* d_out, int out_size, void* d_ws, size_t ws_size,
                              hipStream_t stream) {
  const float* h      = (const float*)d_in[0];
  const float* eig    = (const float*)d_in[1];
  const float* ef     = (const float*)d_in[2];
  const int*   src    = (const int*)d_in[3];
  const int*   dst    = (const int*)d_in[4];
  const float* snorm  = (const float*)d_in[5];
  const float* W_pre  = (const float*)d_in[6];
  const float* b_pre  = (const float*)d_in[7];
  const float* ft_W1  = (const float*)d_in[8];
  const float* ft_b1  = (const float*)d_in[9];
  const float* ft_W2  = (const float*)d_in[10];
  const float* ft_b2  = (const float*)d_in[11];
  const float* ft_W3  = (const float*)d_in[12];
  const float* ft_b3  = (const float*)d_in[13];
  const float* W_post = (const float*)d_in[14];
  const float* b_post = (const float*)d_in[15];
  const float* gamma  = (const float*)d_in[16];
  const float* beta   = (const float*)d_in[17];
  float* out = (float*)d_out;

  char* p = (char*)d_ws;
  size_t off = 0;
  bf16_t* hb     = (bf16_t*)(p + off); off = align_up(off + (size_t)N_NODES * IN_F * 2);
  bf16_t* efb    = (bf16_t*)(p + off); off = align_up(off + (size_t)N_EDGES * ED_F * 2);
  bf16_t* WpreF  = (bf16_t*)(p + off); off = align_up(off + (size_t)9 * 8 * 32 * 16 * 2);
  bf16_t* WpostF = (bf16_t*)(p + off); off = align_up(off + (size_t)64 * 8 * 32 * 16 * 2);
  float*  wl     = (float*)(p + off);  off = align_up(off + (size_t)N_EDGES * 4);
  float*  deg    = (float*)(p + off);  off = align_up(off + (size_t)N_NODES * 4);
  float*  sum    = (float*)(p + off);  off = align_up(off + (size_t)N_NODES * IN_F * 4);
  float*  sumsq  = (float*)(p + off);  off = align_up(off + (size_t)N_NODES * IN_F * 4);
  float*  mx     = (float*)(p + off);  off = align_up(off + (size_t)N_NODES * IN_F * 4);
  float*  mn     = (float*)(p + off);  off = align_up(off + (size_t)N_NODES * IN_F * 4);
  float*  el     = (float*)(p + off);  off = align_up(off + (size_t)N_NODES * IN_F * 4);
  bf16_t* xb     = (bf16_t*)(p + off); off = align_up(off + (size_t)N_NODES * 2048 * 2);
  float*  outpre = (float*)(p + off);  off = align_up(off + (size_t)N_NODES * OUT_F * 4);
  float*  colsum = (float*)(p + off);  off = align_up(off + (size_t)OUT_F * 4);
  float*  colsq  = (float*)(p + off);  off = align_up(off + (size_t)OUT_F * 4);
  (void)ws_size; (void)n_in; (void)in_sizes; (void)out_size;

  const int T = 256;
  int nNF = N_NODES * IN_F;

  k_init<<<(nNF + T - 1) / T, T, 0, stream>>>(sum, sumsq, el, mx, mn, deg, colsum, colsq);
  k_cvt<<<(nNF + T - 1) / T, T, 0, stream>>>(hb, h, nNF);
  k_cvt<<<((N_EDGES * ED_F) + T - 1) / T, T, 0, stream>>>(efb, ef, N_EDGES * ED_F);
  k_wfrag<<<((9 * 8 * 32 * 16) + T - 1) / T, T, 0, stream>>>(WpreF, W_pre, 272, 9);
  k_wfrag<<<((64 * 8 * 32 * 16) + T - 1) / T, T, 0, stream>>>(WpostF, W_post, 2048, 64);
  k_edgefilt<<<(N_EDGES + T - 1) / T, T, 0, stream>>>(eig, src, dst, ft_W1, ft_b1,
                                                      ft_W2, ft_b2, ft_W3, ft_b3, wl, deg);
  k_edge_gemm<<<3750, T, 0, stream>>>(hb, efb, WpreF, b_pre, src, dst, wl,
                                      sum, sumsq, el, mx, mn);
  k_build_xb<<<(nNF + T - 1) / T, T, 0, stream>>>(h, deg, sum, sumsq, mx, mn, el, xb);
  k_node_gemm<<<235, T, 0, stream>>>(xb, WpostF, b_post, snorm, outpre, colsum, colsq);
  k_bn<<<(nNF + T - 1) / T, T, 0, stream>>>(outpre, colsum, colsq, gamma, beta, out);
}
